// SelfAttentionBlock_42202348650993
// MI455X (gfx1250) — matrix operand order repaired and checked by reading, not compiled, not run on hardware
//
#include <hip/hip_runtime.h>
#include <hip/hip_bf16.h>
#include <math.h>

typedef _Float16 f16;
typedef _Float16 v16h __attribute__((ext_vector_type(16)));
typedef float    v8f  __attribute__((ext_vector_type(8)));

#define BATCH   4
#define CCH     256
#define NSP     4096
#define GROUPS_ 8
#define CPG     32
#define THREEC  768
#define EPS_    1e-5f
#define SCALE_  0.0625f   // 1/sqrt(256)

static __device__ __forceinline__ int lane_id() { return threadIdx.x & 31; }

// ---- WMMA f16 16x16x32 fragment loaders (contiguous k order, same for A and B) ------------
// A (16x32): lane L -> row M=L&15, K base = 16*(L>>4); 16 contiguous halves -> 2x b128.
__device__ __forceinline__ v16h load_a16(const f16* __restrict__ src, int ld) {
  const int l = lane_id();
  const int m = l & 15;
  const int hi = l >> 4;
  v16h a;
#pragma unroll
  for (int i = 0; i < 16; ++i) {
    int k = i + hi * 16;
    a[i] = src[m * ld + k];
  }
  return a;
}

// B (32x16) from memory holding B transposed (mem[col][k], k contiguous):
// lane L -> col N=L&15, K base = 16*(L>>4); 16 contiguous halves -> 2x b128.
__device__ __forceinline__ v16h load_bt16(const f16* __restrict__ src, int ld) {
  const int l = lane_id();
  const int n = l & 15;
  const int kb = (l >> 4) * 16;
  v16h b;
#pragma unroll
  for (int i = 0; i < 16; ++i) b[i] = src[n * ld + kb + i];
  return b;
}

// ---- gfx1250 async memory->LDS copy (ASYNCcnt-tracked), per ISA 08 ------
__device__ __forceinline__ void async_to_lds_b128(const f16* gsrc, f16* lds_dst) {
  // low 32 bits of the generic LDS pointer are the LDS byte offset
  unsigned lds_off = (unsigned)(unsigned long long)(void*)lds_dst;
  asm volatile("global_load_async_to_lds_b128 %0, %1, off"
               :: "v"(lds_off), "v"(gsrc) : "memory");
}

// ---- kernel 0: weights f32 -> f16 ---------------------------------------
__global__ void cvt_kernel(const float* __restrict__ qw, const float* __restrict__ pw,
                           f16* __restrict__ qw16, f16* __restrict__ pw16) {
  int i = blockIdx.x * blockDim.x + threadIdx.x;
  if (i < THREEC * CCH) qw16[i] = (f16)qw[i];
  if (i < CCH * CCH)    pw16[i] = (f16)pw[i];
}

// ---- kernel 1: GroupNorm -> h^T in f16 [B][N][C] ------------------------
__global__ void gn_kernel(const float* __restrict__ x, const float* __restrict__ gw,
                          const float* __restrict__ gb, f16* __restrict__ h16t) {
  __shared__ float red[256];
  __shared__ float red2[256];
  const int bg = blockIdx.x;              // b*GROUPS + g
  const int b = bg / GROUPS_, g = bg % GROUPS_;
  const size_t base = (size_t)b * CCH * NSP + (size_t)g * CPG * NSP;
  const int tot = CPG * NSP;              // 131072
  float s = 0.f, s2 = 0.f;
  for (int i = threadIdx.x; i < tot; i += 256) {
    float v = x[base + i];
    s += v; s2 += v * v;
  }
  red[threadIdx.x] = s; red2[threadIdx.x] = s2;
  __syncthreads();
  for (int off = 128; off > 0; off >>= 1) {
    if ((int)threadIdx.x < off) {
      red[threadIdx.x]  += red[threadIdx.x + off];
      red2[threadIdx.x] += red2[threadIdx.x + off];
    }
    __syncthreads();
  }
  const float mean = red[0] / (float)tot;
  const float var  = red2[0] / (float)tot - mean * mean;
  const float rstd = rsqrtf(var + EPS_);
  const int cs = threadIdx.x & 31;        // channel within group
  const int c  = g * CPG + cs;
  const float ga = gw[c], be = gb[c];
  for (int n = (int)threadIdx.x >> 5; n < NSP; n += 8) {
    float v = (x[base + (size_t)cs * NSP + n] - mean) * rstd * ga + be;
    h16t[(size_t)b * NSP * CCH + (size_t)n * CCH + c] = (f16)v;
  }
}

// ---- kernel 2: QKV GEMM -> qt Q^T [B][N][C] (scaled), kt K^T [B][N][C], v16 V [B][C][N]
__global__ void qkv_kernel(const f16* __restrict__ w16, const float* __restrict__ qkvb,
                           const f16* __restrict__ h16t,
                           f16* __restrict__ qt, f16* __restrict__ kt, f16* __restrict__ v16) {
  const int wave = blockIdx.x * 8 + ((int)threadIdx.x >> 5);
  const int b   = wave / (48 * 256);
  const int rem = wave % (48 * 256);
  const int o0  = (rem / 256) * 16;
  const int n0  = (rem % 256) * 16;
  v8f acc = {};
#pragma unroll
  for (int c0 = 0; c0 < CCH; c0 += 32) {
    v16h a  = load_a16(w16 + o0 * CCH + c0, CCH);
    v16h bm = load_bt16(h16t + (size_t)b * NSP * CCH + (size_t)n0 * CCH + c0, CCH);
    acc = __builtin_amdgcn_wmma_f32_16x16x32_f16(false, a, false, bm, (short)0, acc, false, false);
  }
  const int l = lane_id();
  const int nn = n0 + (l & 15);
  const int hf = l >> 4;
#pragma unroll
  for (int r = 0; r < 8; ++r) {
    int o = o0 + r + hf * 8;
    float v = acc[r] + qkvb[o];
    if (o < CCH) {
      qt[(size_t)b * NSP * CCH + (size_t)nn * CCH + o] = (f16)(v * SCALE_);
    } else if (o < 2 * CCH) {
      kt[(size_t)b * NSP * CCH + (size_t)nn * CCH + (o - CCH)] = (f16)v;
    } else {
      v16[(size_t)b * CCH * NSP + (size_t)(o - 2 * CCH) * NSP + nn] = (f16)v;
    }
  }
}

// ---- kernel 3: flash attention; async double-buffered K/V staging -------
__global__ __launch_bounds__(256, 1) void attn_kernel(const f16* __restrict__ qt,
                                                      const f16* __restrict__ kt,
                                                      const f16* __restrict__ v16,
                                                      f16* __restrict__ ot) {
  __shared__ __align__(16) f16 lds_k[2][32 * CCH];  // K^T chunk [m_local][c]  2x16KB
  __shared__ __align__(16) f16 lds_v[2][CCH * 32];  // V   chunk [c][m_local]  2x16KB
  __shared__ __align__(16) f16 lds_p[8][16 * 32];   // per-wave P tile         8KB
  const int wid  = (int)threadIdx.x >> 5;
  const int wave = blockIdx.x * 8 + wid;
  const int b  = wave >> 8;               // uniform per block
  const int n0 = (wave & 255) * 16;
  const int l  = lane_id();
  const int m_ = l & 15;
  const int hf = l >> 4;
  const int t  = (int)threadIdx.x;

  const f16* ktb = kt  + (size_t)b * NSP * CCH;
  const f16* vbb = v16 + (size_t)b * CCH * NSP;

  // preload Q^T tile as 8 A-fragments (vectorized b128 loads)
  v16h aq[8];
#pragma unroll
  for (int cc = 0; cc < 8; ++cc)
    aq[cc] = load_a16(qt + (size_t)b * NSP * CCH + (size_t)n0 * CCH + cc * 32, CCH);

  v8f zero = {};
  v8f oacc[16];
#pragma unroll
  for (int cc = 0; cc < 16; ++cc) oacc[cc] = zero;

  float mrun[8], lrun[8];
#pragma unroll
  for (int r = 0; r < 8; ++r) { mrun[r] = -__builtin_inff(); lrun[r] = 0.f; }

  // async staging of one 32-key chunk: 4x K + 4x V b128 per thread (8 ASYNC ops)
  auto stage = [&](int bufsel, int mbase) {
#pragma unroll
    for (int j = 0; j < 4; ++j) {         // K^T: 32 rows x 512B
      int ci = t + j * 256;
      int row = ci >> 5, off = (ci & 31) * 8;
      async_to_lds_b128(ktb + (size_t)(mbase + row) * CCH + off,
                        &lds_k[bufsel][row * CCH + off]);
    }
#pragma unroll
    for (int j = 0; j < 4; ++j) {         // V: 256 rows x 64B
      int ci = t + j * 256;
      int c = ci >> 2, off = (ci & 3) * 8;
      async_to_lds_b128(vbb + (size_t)c * NSP + mbase + off,
                        &lds_v[bufsel][c * 32 + off]);
    }
  };

  stage(0, 0);                            // prologue: chunk 0 -> buf 0
  int buf = 0;

  for (int m0 = 0; m0 < NSP; m0 += 32) {
    if (m0 + 32 < NSP) {                  // issue next chunk into other buffer
      stage(buf ^ 1, m0 + 32);
      asm volatile("s_wait_asynccnt 0x8" ::: "memory");   // current chunk resident
    } else {
      asm volatile("s_wait_asynccnt 0x0" ::: "memory");
    }
    __syncthreads();                      // all waves' copies of current chunk done

    const f16* lk = lds_k[buf];
    const f16* lv = lds_v[buf];
    const int kb = hf * 16;

    // S tiles (cols m0..+15 and m0+16..+31); contiguous LDS fragment reads
    v8f s0 = zero, s1 = zero;
#pragma unroll
    for (int cc = 0; cc < 8; ++cc) {
      v16h bk0, bk1;
#pragma unroll
      for (int i = 0; i < 16; ++i) bk0[i] = lk[m_ * CCH + cc * 32 + kb + i];
#pragma unroll
      for (int i = 0; i < 16; ++i) bk1[i] = lk[(16 + m_) * CCH + cc * 32 + kb + i];
      s0 = __builtin_amdgcn_wmma_f32_16x16x32_f16(false, aq[cc], false, bk0, (short)0, s0, false, false);
      s1 = __builtin_amdgcn_wmma_f32_16x16x32_f16(false, aq[cc], false, bk1, (short)0, s1, false, false);
    }
    // online softmax per row (rows 0-7 on lanes 0-15, rows 8-15 on lanes 16-31)
#pragma unroll
    for (int r = 0; r < 8; ++r) {
      float v0 = s0[r], v1 = s1[r];
      float mx = fmaxf(v0, v1);
#pragma unroll
      for (int off = 1; off < 16; off <<= 1) mx = fmaxf(mx, __shfl_xor(mx, off, 32));
      float mnew = fmaxf(mrun[r], mx);
      float p0 = __expf(v0 - mnew);
      float p1 = __expf(v1 - mnew);
      float rs = p0 + p1;
#pragma unroll
      for (int off = 1; off < 16; off <<= 1) rs += __shfl_xor(rs, off, 32);
      float alpha = __expf(mrun[r] - mnew);
      lrun[r] = lrun[r] * alpha + rs;
      mrun[r] = mnew;
      int row = r + hf * 8;
      lds_p[wid][row * 32 + m_]      = (f16)p0;
      lds_p[wid][row * 32 + 16 + m_] = (f16)p1;
#pragma unroll
      for (int cc = 0; cc < 16; ++cc) oacc[cc][r] *= alpha;
    }
    // P tile back in A-layout (wave-private LDS region, in-order DS)
    v16h pa;
#pragma unroll
    for (int i = 0; i < 16; ++i) {
      int k = i + hf * 16;
      pa[i] = lds_p[wid][m_ * 32 + k];
    }
    // O^T += P x V  (B-fragment: mem[col=c][k=m] contiguous from lds_v)
#pragma unroll
    for (int cc = 0; cc < 16; ++cc) {
      v16h bv;
#pragma unroll
      for (int i = 0; i < 16; ++i) bv[i] = lv[(cc * 16 + m_) * 32 + kb + i];
      oacc[cc] = __builtin_amdgcn_wmma_f32_16x16x32_f16(false, pa, false, bv, (short)0, oacc[cc], false, false);
    }
    __syncthreads();                      // buffer consumed before refill next iter
    buf ^= 1;
  }
  // finalize: divide by row sums, write O^T [B][N][C] f16
#pragma unroll
  for (int r = 0; r < 8; ++r) {
    float inv = 1.f / lrun[r];
    int n = n0 + r + hf * 8;
#pragma unroll
    for (int cc = 0; cc < 16; ++cc) {
      ot[(size_t)b * NSP * CCH + (size_t)n * CCH + cc * 16 + m_] = (f16)(oacc[cc][r] * inv);
    }
  }
}

// ---- kernel 4: proj: out[o][n] = sum_c Wp[o][c] * O[c][n] + bias + x ----
__global__ void proj_kernel(const float* __restrict__ x, const f16* __restrict__ wp,
                            const float* __restrict__ pb, const f16* __restrict__ ot,
                            float* __restrict__ out) {
  const int wave = blockIdx.x * 8 + ((int)threadIdx.x >> 5);
  const int b   = wave / (16 * 256);
  const int rem = wave % (16 * 256);
  const int o0  = (rem / 256) * 16;
  const int n0  = (rem % 256) * 16;
  v8f acc = {};
#pragma unroll
  for (int c0 = 0; c0 < CCH; c0 += 32) {
    v16h a  = load_a16(wp + o0 * CCH + c0, CCH);
    v16h bm = load_bt16(ot + (size_t)b * NSP * CCH + (size_t)n0 * CCH + c0, CCH);
    acc = __builtin_amdgcn_wmma_f32_16x16x32_f16(false, a, false, bm, (short)0, acc, false, false);
  }
  const int l  = lane_id();
  const int nn = n0 + (l & 15);
  const int hf = l >> 4;
#pragma unroll
  for (int r = 0; r < 8; ++r) {
    int o = o0 + r + hf * 8;
    size_t idx = (size_t)b * CCH * NSP + (size_t)o * NSP + nn;
    out[idx] = x[idx] + acc[r] + pb[o];   // 64B-coalesced per half-wave row
  }
}

extern "C" void kernel_launch(void* const* d_in, const int* in_sizes, int n_in,
                              void* d_out, int out_size, void* d_ws, size_t ws_size,
                              hipStream_t stream) {
  const float* x     = (const float*)d_in[0];
  const float* gnw   = (const float*)d_in[1];
  const float* gnb   = (const float*)d_in[2];
  const float* qkvw  = (const float*)d_in[3];
  const float* qkvb  = (const float*)d_in[4];
  const float* projw = (const float*)d_in[5];
  const float* projb = (const float*)d_in[6];
  float* out = (float*)d_out;

  char* p = (char*)d_ws;
  f16* w16q = (f16*)p; p += (size_t)THREEC * CCH * 2;
  f16* w16p = (f16*)p; p += (size_t)CCH * CCH * 2;
  f16* h16t = (f16*)p; p += (size_t)BATCH * NSP * CCH * 2;
  f16* qt   = (f16*)p; p += (size_t)BATCH * NSP * CCH * 2;
  f16* kt   = (f16*)p; p += (size_t)BATCH * NSP * CCH * 2;
  f16* v16  = (f16*)p; p += (size_t)BATCH * CCH * NSP * 2;
  f16* ot   = (f16*)p; p += (size_t)BATCH * NSP * CCH * 2;

  cvt_kernel<<<768, 256, 0, stream>>>(qkvw, projw, w16q, w16p);
  gn_kernel<<<BATCH * GROUPS_, 256, 0, stream>>>(x, gnw, gnb, h16t);
  qkv_kernel<<<6144, 256, 0, stream>>>(w16q, qkvb, h16t, qt, kt, v16);
  attn_kernel<<<128, 256, 0, stream>>>(qt, kt, v16, ot);
  proj_kernel<<<2048, 256, 0, stream>>>(x, w16p, projb, ot, out);
}
